// self_attention_block_84602265796708
// MI455X (gfx1250) — matrix operand order repaired and checked by reading, not compiled, not run on hardware
//
#include <hip/hip_runtime.h>
#include <hip/hip_bf16.h>
#include <stdint.h>

// ---------------------------------------------------------------------------
// MHA block for MI455X (gfx1250): bf16 WMMA (v_wmma_f32_16x16x32_bf16),
// f32 accumulate, flash-attention with block-cooperative double-buffered
// async LDS staging (GLOBAL_LOAD_ASYNC_TO_LDS_B128 + s_wait_asynccnt).
// ---------------------------------------------------------------------------

typedef __bf16 bf16_t;
typedef __attribute__((ext_vector_type(16))) __bf16 v16bf;
typedef __attribute__((ext_vector_type(8)))  __bf16 v8bf;
typedef __attribute__((ext_vector_type(8)))  float  v8f;

static constexpr int Bn = 4;
static constexpr int Sn = 2048;
static constexpr int En = 128;   // per-head dim
static constexpr int Hn = 8;
static constexpr int Mn = Bn * Sn;        // 8192 token rows
static constexpr int Nqkv = Hn * En;      // 1024

__device__ __forceinline__ v8f zero8() {
  v8f z;
#pragma unroll
  for (int i = 0; i < 8; ++i) z[i] = 0.0f;
  return z;
}

// A-fragment (16x32 bf16, row-major source, leading dim ld):
// lane<16: row=lane, K = {0..7} U {16..23}; lane>=16: row=lane-16, K = {8..15} U {24..31}
__device__ __forceinline__ v16bf load_a_frag(const bf16_t* p, int ld) {
  const int lane = threadIdx.x & 31;
  const int row = lane & 15;
  const int hi  = lane >> 4;
  const bf16_t* r0 = p + row * ld + hi * 8;
  v8bf c0 = *(const v8bf*)(r0);
  v8bf c1 = *(const v8bf*)(r0 + 16);
  v16bf a;
#pragma unroll
  for (int i = 0; i < 8; ++i) { a[i] = c0[i]; a[i + 8] = c1[i]; }
  return a;
}

// B-fragment (32x16 bf16) loaded from a K-contiguous ("B-transposed") layout:
// source is [N][K] row-major; p points at (n0,k0); lane column n=lane&15,
// K = {0..7} U {16..23} for lane<16, {8..15} U {24..31} for lane>=16.
__device__ __forceinline__ v16bf load_b_frag(const bf16_t* p, int ld) {
  const int lane = threadIdx.x & 31;
  const int n  = lane & 15;
  const int hi = lane >> 4;
  const bf16_t* r0 = p + n * ld + hi * 8;
  v8bf c0 = *(const v8bf*)(r0);
  v8bf c1 = *(const v8bf*)(r0 + 16);
  v16bf b;
#pragma unroll
  for (int i = 0; i < 8; ++i) { b[i] = c0[i]; b[i + 8] = c1[i]; }
  return b;
}

__device__ __forceinline__ v8f wmma_bf16(v16bf a, v16bf b, v8f c) {
  return __builtin_amdgcn_wmma_f32_16x16x32_bf16(false, a, false, b, (short)0, c,
                                                 false, false);
}

// C-fragment (16x16 f32) -> bf16 row-major store: lane col n, rows hi*8+r.
__device__ __forceinline__ void store_c_bf16(bf16_t* p, int ld, v8f c) {
  const int lane = threadIdx.x & 31;
  const int n  = lane & 15;
  const int hi = lane >> 4;
#pragma unroll
  for (int r = 0; r < 8; ++r) p[(hi * 8 + r) * ld + n] = (bf16_t)c[r];
}

// ---------------------------------------------------------------------------
// gfx1250 async copy: global -> LDS, 16 bytes per lane, tracked by ASYNCcnt.
// lds_addr = low 32 bits of the generic pointer (flat->LDS truncation).
// ---------------------------------------------------------------------------
__device__ __forceinline__ void async_load_b128(uint32_t lds_addr,
                                                const bf16_t* gptr) {
  asm volatile("global_load_async_to_lds_b128 %0, %1, off"
               :: "v"(lds_addr), "v"(gptr)
               : "memory");
}

__device__ __forceinline__ void wait_asynccnt_4() {
#if __has_builtin(__builtin_amdgcn_s_wait_asynccnt)
  __builtin_amdgcn_s_wait_asynccnt(4);
#else
  asm volatile("s_wait_asynccnt 0x4" ::: "memory");
#endif
}
__device__ __forceinline__ void wait_asynccnt_0() {
#if __has_builtin(__builtin_amdgcn_s_wait_asynccnt)
  __builtin_amdgcn_s_wait_asynccnt(0);
#else
  asm volatile("s_wait_asynccnt 0x0" ::: "memory");
#endif
}

// ---------------------------------------------------------------------------
// Stage 0: precision conversion / layout transforms
// ---------------------------------------------------------------------------
__global__ void cvt_f32_bf16_kernel(const float* __restrict__ src,
                                    bf16_t* __restrict__ dst, int n) {
  int i = blockIdx.x * blockDim.x + threadIdx.x;
  if (i < n) dst[i] = (bf16_t)src[i];
}

// src [R][C] f32 -> dst [C][R] bf16 (K-contiguous weights for B-fragments)
__global__ void cvt_transpose_kernel(const float* __restrict__ src,
                                     bf16_t* __restrict__ dst, int R, int C) {
  int i = blockIdx.x * blockDim.x + threadIdx.x;
  if (i < R * C) {
    int r = i / C, c = i - r * C;
    dst[c * R + r] = (bf16_t)src[i];
  }
}

// ---------------------------------------------------------------------------
// Stage 1: QKV projection. X[8192][128]bf16 @ W[128][1024] (via WT[1024][128]).
// One wave computes a 16x64 tile (A reused over 4 B-fragments).
// blockIdx.y selects {Q,K,V}. Q,K stored [B,H,S,128]; V stored [B,H,128,S].
// ---------------------------------------------------------------------------
__global__ __launch_bounds__(256) void qkv_gemm_kernel(
    const bf16_t* __restrict__ X, const bf16_t* __restrict__ WTall,
    bf16_t* __restrict__ Q, bf16_t* __restrict__ K, bf16_t* __restrict__ VT) {
  const int wave = threadIdx.x >> 5;
  const int lane = threadIdx.x & 31;
  const int tid  = blockIdx.x * 8 + wave;     // 8192 tiles
  const int mt = tid >> 4;                    // [0,512)
  const int nt = tid & 15;                    // [0,16) : 64-wide column block
  const int row0 = mt * 16;
  const int col0 = nt * 64;
  const int which = blockIdx.y;
  const bf16_t* WT = WTall + which * (Nqkv * En);

  v8f acc[4] = {zero8(), zero8(), zero8(), zero8()};
#pragma unroll
  for (int kk = 0; kk < En; kk += 32) {
    v16bf a = load_a_frag(X + row0 * En + kk, En);
#pragma unroll
    for (int j = 0; j < 4; ++j) {
      v16bf b = load_b_frag(WT + (col0 + j * 16) * En + kk, En);
      acc[j] = wmma_bf16(a, b, acc[j]);
    }
  }

  const int b = row0 / Sn;
  const int s = row0 - b * Sn;
  const int n  = lane & 15;
  const int hi = lane >> 4;
#pragma unroll
  for (int j = 0; j < 4; ++j) {
    const int cg = col0 + j * 16;      // global col in [0,1024)
    const int h  = cg >> 7;
    const int d0 = cg & 127;
    if (which < 2) {
      bf16_t* dst = (which == 0 ? Q : K) + ((b * Hn + h) * Sn + s) * En + d0;
      store_c_bf16(dst, En, acc[j]);
    } else {
      // VT[((b*H+h)*128 + d)*S + s] : element (row=s+hi*8+r, col=d0+n)
      bf16_t* dst = VT + ((b * Hn + h) * En + d0) * Sn + s;
#pragma unroll
      for (int r = 0; r < 8; ++r)
        dst[n * Sn + (hi * 8 + r)] = (bf16_t)acc[j][r];
    }
  }
}

// ---------------------------------------------------------------------------
// Stage 2: flash attention. All 8 waves of a block share one (b,h) and stream
// the same K/V tiles: double-buffered async global->LDS staging, WMMA from LDS.
// ---------------------------------------------------------------------------
__global__ __launch_bounds__(256) void attn_kernel(
    const bf16_t* __restrict__ Q, const bf16_t* __restrict__ K,
    const bf16_t* __restrict__ VT, bf16_t* __restrict__ O) {
  __shared__ bf16_t Kbuf[2][32 * 128];  // [t][d] tile, 8 KB per buffer
  __shared__ bf16_t Vbuf[2][128 * 32];  // [d][t] tile, 8 KB per buffer
  __shared__ bf16_t Pbuf[8][16 * 32];   // wave-private P staging (C->A relayout)

  const int wave = threadIdx.x >> 5;
  const int lane = threadIdx.x & 31;
  const int tid  = threadIdx.x;

  // 512 blocks: bh = blockIdx.x/16 in [0,32); 16 blocks cover S rows of one bh.
  const int bh   = blockIdx.x >> 4;
  const int row0 = ((blockIdx.x & 15) * 8 + wave) * 16;

  const bf16_t* Qp = Q + (size_t)bh * Sn * En + row0 * En;
  const bf16_t* Kp = K + (size_t)bh * Sn * En;
  const bf16_t* Vp = VT + (size_t)bh * En * Sn;
  bf16_t* Op = O + (size_t)bh * Sn * En + row0 * En;

  v16bf qa[4];
#pragma unroll
  for (int kk = 0; kk < 4; ++kk) qa[kk] = load_a_frag(Qp + kk * 32, En);

  v8f o[8];
  float m[8], l[8];
#pragma unroll
  for (int r = 0; r < 8; ++r) { m[r] = -3.0e38f; l[r] = 0.0f; }
#pragma unroll
  for (int f = 0; f < 8; ++f) o[f] = zero8();

  const float scale = 0.08838834764831845f;   // 1/sqrt(128)
  const int n  = lane & 15;
  const int hi = lane >> 4;
  bf16_t* pl = &Pbuf[wave][0];

  // Fill one 32-key double buffer slot: K tile 32x128, V tile 128x32.
  // 256 threads x (2+2) b128 transfers = 16 KB; 4 async ops per wave per fill.
  auto issue_fill = [&](int buf, int t0) {
#pragma unroll
    for (int i = 0; i < 2; ++i) {
      const int e = tid * 2 + i;                 // [0,512)
      const int trow = e >> 4, ck = e & 15;
      async_load_b128(
          (uint32_t)(uintptr_t)(&Kbuf[buf][trow * 128 + ck * 8]),
          Kp + (t0 + trow) * En + ck * 8);
    }
#pragma unroll
    for (int i = 0; i < 2; ++i) {
      const int e = tid * 2 + i;                 // [0,512)
      const int drow = e >> 2, ck = e & 3;
      async_load_b128(
          (uint32_t)(uintptr_t)(&Vbuf[buf][drow * 32 + ck * 8]),
          Vp + drow * Sn + t0 + ck * 8);
    }
  };

  constexpr int NIT = Sn / 32;                   // 64 key-chunks
  issue_fill(0, 0);

  for (int it = 0; it < NIT; ++it) {
    const int buf = it & 1;
    if (it + 1 < NIT) issue_fill(buf ^ 1, (it + 1) * 32);
    // Async loads complete in order: <=4 outstanding means current buf landed.
    if (it + 1 < NIT) wait_asynccnt_4(); else wait_asynccnt_0();
    __syncthreads();                             // all waves' fills visible

    const bf16_t* Kt = &Kbuf[buf][0];
    const bf16_t* Vt = &Vbuf[buf][0];

    // --- scores S[16 x 32] = Q(16x128) @ K^T, K fragments from LDS ---
    v8f s0 = zero8(), s1 = zero8();
#pragma unroll
    for (int kk = 0; kk < 4; ++kk) {
      v16bf b0 = load_b_frag(Kt + kk * 32, 128);
      v16bf b1 = load_b_frag(Kt + 16 * 128 + kk * 32, 128);
      s0 = wmma_bf16(qa[kk], b0, s0);
      s1 = wmma_bf16(qa[kk], b1, s1);
    }
    // --- online softmax over this 32-key chunk (row = VGPR index, per half) ---
#pragma unroll
    for (int r = 0; r < 8; ++r) {
      float v0 = s0[r] * scale, v1 = s1[r] * scale;
      float mx = fmaxf(v0, v1);
      mx = fmaxf(mx, __shfl_xor(mx, 1, 16));
      mx = fmaxf(mx, __shfl_xor(mx, 2, 16));
      mx = fmaxf(mx, __shfl_xor(mx, 4, 16));
      mx = fmaxf(mx, __shfl_xor(mx, 8, 16));
      const float mn = fmaxf(m[r], mx);
      const float alpha = __expf(m[r] - mn);
      m[r] = mn;
      const float p0 = __expf(v0 - mn);
      const float p1 = __expf(v1 - mn);
      s0[r] = p0; s1[r] = p1;
      float rs = p0 + p1;
      rs += __shfl_xor(rs, 1, 16);
      rs += __shfl_xor(rs, 2, 16);
      rs += __shfl_xor(rs, 4, 16);
      rs += __shfl_xor(rs, 8, 16);
      l[r] = l[r] * alpha + rs;
#pragma unroll
      for (int f = 0; f < 8; ++f) o[f][r] *= alpha;
    }
    // --- relayout P (C-fragment) -> A-fragment via wave-private LDS tile ---
#pragma unroll
    for (int r = 0; r < 8; ++r) {
      pl[(hi * 8 + r) * 32 + n]      = (bf16_t)s0[r];
      pl[(hi * 8 + r) * 32 + 16 + n] = (bf16_t)s1[r];
    }
    v16bf pa = load_a_frag(pl, 32);
    // --- O += P(16x32) @ V(32x128), V fragments from LDS [d][t] tile ---
#pragma unroll
    for (int f = 0; f < 8; ++f) {
      v16bf vb = load_b_frag(Vt + (f * 16) * 32, 32);
      o[f] = wmma_bf16(pa, vb, o[f]);
    }
    __syncthreads();   // everyone done with `buf` before it is refilled
  }

#pragma unroll
  for (int r = 0; r < 8; ++r) {
    const float inv = 1.0f / l[r];
#pragma unroll
    for (int f = 0; f < 8; ++f) o[f][r] *= inv;
  }
#pragma unroll
  for (int f = 0; f < 8; ++f) store_c_bf16(Op + f * 16, En, o[f]);
}

// ---------------------------------------------------------------------------
// Stage 3: output projection. O[B,H,S,128] (A, h selected per K-tile) @
// Wf[1024][128] (via WfT[128][1024] B-fragments) + bias -> f32 out [B,S,128].
// ---------------------------------------------------------------------------
__global__ __launch_bounds__(256) void proj_kernel(
    const bf16_t* __restrict__ O, const bf16_t* __restrict__ WfT,
    const float* __restrict__ bias, float* __restrict__ out) {
  const int wave = threadIdx.x >> 5;
  const int lane = threadIdx.x & 31;
  const int tid  = blockIdx.x * 8 + wave;     // 4096 tiles = 512 x 8
  const int mt = tid >> 3;
  const int nt = tid & 7;
  const int row0 = mt * 16;
  const int col0 = nt * 16;
  const int b = row0 / Sn;
  const int srow = row0 - b * Sn;

  v8f acc = zero8();
#pragma unroll 4
  for (int kt = 0; kt < Nqkv; kt += 32) {
    const int h  = kt >> 7;
    const int d0 = kt & 127;
    v16bf a = load_a_frag(O + ((size_t)(b * Hn + h) * Sn + srow) * En + d0, En);
    v16bf w = load_b_frag(WfT + col0 * Nqkv + kt, Nqkv);
    acc = wmma_bf16(a, w, acc);
  }

  const int n  = lane & 15;
  const int hi = lane >> 4;
  const float bv = bias[col0 + n];
  float* dst = out + (size_t)row0 * En + col0;
#pragma unroll
  for (int r = 0; r < 8; ++r) dst[(hi * 8 + r) * En + n] = acc[r] + bv;
}

// ---------------------------------------------------------------------------
extern "C" void kernel_launch(void* const* d_in, const int* in_sizes, int n_in,
                              void* d_out, int out_size, void* d_ws, size_t ws_size,
                              hipStream_t stream) {
  const float* x  = (const float*)d_in[0];
  const float* Wq = (const float*)d_in[1];
  const float* Wk = (const float*)d_in[2];
  const float* Wv = (const float*)d_in[3];
  const float* Wf = (const float*)d_in[4];
  const float* bf = (const float*)d_in[5];
  float* out = (float*)d_out;

  // Workspace carve-up (bf16 elements). Total ~70 MB.
  bf16_t* ws = (bf16_t*)d_ws;
  size_t off = 0;
  bf16_t* Xb  = ws + off; off += (size_t)Mn * En;          // 1,048,576
  bf16_t* WqT = ws + off; off += (size_t)Nqkv * En;        // 131,072
  bf16_t* WkT = ws + off; off += (size_t)Nqkv * En;
  bf16_t* WvT = ws + off; off += (size_t)Nqkv * En;
  bf16_t* WfT = ws + off; off += (size_t)En * Nqkv;
  bf16_t* Qb  = ws + off; off += (size_t)Bn * Hn * Sn * En; // 8,388,608 each
  bf16_t* Kb  = ws + off; off += (size_t)Bn * Hn * Sn * En;
  bf16_t* VTb = ws + off; off += (size_t)Bn * Hn * En * Sn;
  bf16_t* Ob  = ws + off; off += (size_t)Bn * Hn * Sn * En;

  // Stage 0: conversions
  cvt_f32_bf16_kernel<<<(Mn * En) / 256, 256, 0, stream>>>(x, Xb, Mn * En);
  cvt_transpose_kernel<<<(En * Nqkv) / 256, 256, 0, stream>>>(Wq, WqT, En, Nqkv);
  cvt_transpose_kernel<<<(En * Nqkv) / 256, 256, 0, stream>>>(Wk, WkT, En, Nqkv);
  cvt_transpose_kernel<<<(En * Nqkv) / 256, 256, 0, stream>>>(Wv, WvT, En, Nqkv);
  cvt_transpose_kernel<<<(Nqkv * En) / 256, 256, 0, stream>>>(Wf, WfT, Nqkv, En);

  // Stage 1: QKV GEMMs (blockIdx.y = which of Q/K/V); 8192 wave-tiles each.
  qkv_gemm_kernel<<<dim3(1024, 3), 256, 0, stream>>>(Xb, WqT, Qb, Kb, VTb);

  // Stage 2: flash attention; 512 blocks, 8 waves sharing one (b,h) K/V stream.
  attn_kernel<<<512, 256, 0, stream>>>(Qb, Kb, VTb, Ob);

  // Stage 3: output projection; 4096 wave-tiles.
  proj_kernel<<<512, 256, 0, stream>>>(Ob, WfT, bf, out);
}
